// CSA_DGCN_38087769981513
// MI455X (gfx1250) — matrix operand order repaired and checked by reading, not compiled, not run on hardware
//
#include <hip/hip_runtime.h>
#include <hip/hip_bf16.h>

// ---------------- constants (match reference) ----------------
#define BB   256
#define NN   128
#define INF_ 9
#define HD   128
#define NHD  8
#define LL   4
#define OUTD 60
#define DHD  16
#define MM   (BB * NN)          // 32768 rows
#define ALPHA_ 0.6f
#define EPS_   1e-5f

typedef __bf16 bf16;
typedef __attribute__((ext_vector_type(16))) __bf16 v16bf;
typedef __attribute__((ext_vector_type(8)))  __bf16 v8bf;
typedef __attribute__((ext_vector_type(8)))  float  v8f;

__device__ __forceinline__ v16bf combine16(v8bf lo, v8bf hi) {
  return __builtin_shufflevector(lo, hi, 0, 1, 2, 3, 4, 5, 6, 7,
                                 8, 9, 10, 11, 12, 13, 14, 15);
}

// A-matrix fragment (16-bit 16x32, row-major source, ld = leading dim):
//   elem j<8 -> K = kbase + half*8 + j ; elem j>=8 -> K = kbase + 16 + half*8 + (j-8)
__device__ __forceinline__ v16bf load_fragA(const bf16* __restrict__ A, int ld,
                                            int row_base, int kbase, int lane) {
  int m    = row_base + (lane & 15);
  int half = lane >> 4;
  const bf16* p = A + (size_t)m * ld + kbase + half * 8;
  v8bf lo = *(const v8bf*)(p);
  v8bf hi = *(const v8bf*)(p + 16);
  return combine16(lo, hi);
}

// B-matrix fragment from PACKED layout: P[((k>>5)*Nn + n)*32 + (k&31)]
__device__ __forceinline__ v16bf load_fragB_packed(const bf16* __restrict__ P, int Nn,
                                                   int col_base, int kbase, int lane) {
  int n    = col_base + (lane & 15);
  int half = lane >> 4;
  const bf16* p = P + ((size_t)(kbase >> 5) * Nn + n) * 32 + half * 8;
  v8bf lo = *(const v8bf*)(p);
  v8bf hi = *(const v8bf*)(p + 16);
  return combine16(lo, hi);
}

// ---------------- prep kernels ----------------

// pack a row-major fp32 [K][Nn] weight into bf16 WMMA-B fragment layout
__global__ void pack_b_kernel(const float* __restrict__ src, bf16* __restrict__ dst,
                              int K, int Nn) {
  int total = K * Nn;
  for (int idx = blockIdx.x * blockDim.x + threadIdx.x; idx < total;
       idx += gridDim.x * blockDim.x) {
    int k = idx / Nn, n = idx - k * Nn;
    dst[((size_t)(k >> 5) * Nn + n) * 32 + (k & 31)] = (bf16)src[idx];
  }
}

// Aw = row_normalize(sigmoid(weak_param) * weak_mask)  -> row-major bf16 (A operand)
__global__ void prep_aw_kernel(const float* __restrict__ wp, const float* __restrict__ wm,
                               bf16* __restrict__ awb) {
  __shared__ float red[NN];
  int i = blockIdx.x, j = threadIdx.x;
  float v = (1.f / (1.f + expf(-wp[i * NN + j]))) * wm[i * NN + j];
  red[j] = v;
  __syncthreads();
  for (int s = 64; s > 0; s >>= 1) { if (j < s) red[j] += red[j + s]; __syncthreads(); }
  awb[i * NN + j] = (bf16)(v / (red[0] + EPS_));
}

// embedding: h = relu(BN(x @ conv_w^T + conv_b))
__global__ void embed_kernel(const float* __restrict__ x, const float* __restrict__ cw,
                             const float* __restrict__ cb, const float* __restrict__ bg,
                             const float* __restrict__ bb2, const float* __restrict__ bm,
                             const float* __restrict__ bv, float* __restrict__ h,
                             bf16* __restrict__ hb) {
  int idx = blockIdx.x * blockDim.x + threadIdx.x;
  if (idx >= MM * HD) return;
  int o  = idx & (HD - 1);
  int bn = idx >> 7;
  const float* xr = x + (size_t)bn * INF_;
  const float* wr = cw + (size_t)o * INF_;
  float acc = cb[o];
#pragma unroll
  for (int i = 0; i < INF_; ++i) acc += xr[i] * wr[i];
  acc = (acc - bm[o]) * rsqrtf(bv[o] + EPS_) * bg[o] + bb2[o];
  acc = fmaxf(acc, 0.f);
  h[idx]  = acc;
  hb[idx] = (bf16)acc;
}

// ---------------- generic WMMA GEMM (compile-time shapes) ----------------
// C[MM,Nn] = A[MM,K](row-major bf16) @ Bp(K x Nn, packed bf16)
// Wave: one 16-row tile x NT 16-col tiles; K-loop fully unrolled.
template <int K, int Nn, int NT, bool HAS_BIAS, bool HAS_RES, bool RELU,
          bool OUT_F32, bool OUT_BF16, bool OUT_BPACK>
__global__ __launch_bounds__(256)
void gemm_kernel(const bf16* __restrict__ A, const bf16* __restrict__ Bp,
                 const float* __restrict__ bias, const float* __restrict__ res,
                 float* __restrict__ of32, bf16* __restrict__ ob16,
                 bf16* __restrict__ obp) {
  int wave = threadIdx.x >> 5;
  int lane = threadIdx.x & 31;
  int mt  = blockIdx.x * 8 + wave;
  int nt0 = blockIdx.y * NT;
  int half = lane >> 4;
  int nl   = lane & 15;

  v8f acc[NT];
#pragma unroll
  for (int q = 0; q < NT; ++q) { v8f z = {}; acc[q] = z; }

  const bf16* arow = A + (size_t)(mt * 16 + nl) * K;
#pragma unroll
  for (int k = 0; k < K; k += 32) {
    const bf16* ap = arow + k + half * 8;
    v16bf a = combine16(*(const v8bf*)ap, *(const v8bf*)(ap + 16));
#pragma unroll
    for (int q = 0; q < NT; ++q) {
      int n = (nt0 + q) * 16 + nl;
      const bf16* bp = Bp + ((size_t)(k >> 5) * Nn + n) * 32 + half * 8;
      v16bf b = combine16(*(const v8bf*)bp, *(const v8bf*)(bp + 16));
      acc[q] = __builtin_amdgcn_wmma_f32_16x16x32_bf16(false, a, false, b, (short)0,
                                                       acc[q], false, false);
    }
  }

#pragma unroll
  for (int q = 0; q < NT; ++q) {
    int n = (nt0 + q) * 16 + nl;
    float bv = HAS_BIAS ? bias[n] : 0.f;
    float vals[8];
#pragma unroll
    for (int v = 0; v < 8; ++v) {
      int m = mt * 16 + v + 8 * half;
      float val = acc[q][v] + bv;
      if (HAS_RES) val += res[(size_t)m * Nn + n];
      if (RELU) val = fmaxf(val, 0.f);
      vals[v] = val;
      if (OUT_F32)  of32[(size_t)m * Nn + n]  = val;
      if (OUT_BF16) ob16[(size_t)m * Nn + n] = (bf16)val;
    }
    if (OUT_BPACK) {
      // per-batch packed-B layout of the output (K index = row within batch)
      int bidx    = (mt * 16) / NN;
      int klocal0 = (mt * 16) & (NN - 1);
      int kb      = klocal0 >> 5;
      v8bf pk;
#pragma unroll
      for (int v = 0; v < 8; ++v) pk[v] = (bf16)vals[v];
      *(v8bf*)(obp + (size_t)bidx * NN * HD + ((size_t)kb * HD + n) * 32 +
               (klocal0 & 16) + 8 * half) = pk;
    }
  }
}

// s/t head projections from hp (fp32)
__global__ void st_kernel(const float* __restrict__ hp, const float* __restrict__ asrc,
                          const float* __restrict__ adst, float* __restrict__ s,
                          float* __restrict__ t) {
  int idx = blockIdx.x * blockDim.x + threadIdx.x;
  if (idx >= MM * NHD) return;
  int hh = idx & (NHD - 1);
  int bn = idx >> 3;
  const float* row = hp + (size_t)bn * HD + hh * DHD;
  float ss = 0.f, tt = 0.f;
#pragma unroll
  for (int d = 0; d < DHD; ++d) {
    ss += row[d] * asrc[hh * DHD + d];
    tt += row[d] * adst[hh * DHD + d];
  }
  s[idx] = ss;
  t[idx] = tt;
}

// attention (per batch b, head h): masked softmax (z computed once via fp32 LDS
// buffer, t-row staged in LDS) -> att(bf16, LDS) -> WMMA aggregation.
__global__ __launch_bounds__(256)
void attn_strong_kernel(const float* __restrict__ s, const float* __restrict__ t,
                        const float* __restrict__ ewa, const float* __restrict__ sadj,
                        const bf16* __restrict__ hpPack, float* __restrict__ strong) {
  __shared__ __align__(16) float ebuf[NN][NN];  // 64 KB
  __shared__ __align__(16) bf16 attb[NN][NN];   // 32 KB
  __shared__ float tvec[NN];
  int b = blockIdx.x, h = blockIdx.y;
  int tid = threadIdx.x;
  if (tid < NN) tvec[tid] = t[((size_t)b * NN + tid) * NHD + h];
  __syncthreads();
  if (tid < NN) {
    int i = tid;
    float sv = s[((size_t)b * NN + i) * NHD + h];
    const float* ewaR = ewa + i * NN;
    const float* sadR = sadj + i * NN;
    float mx = -3.4e38f;
    for (int j = 0; j < NN; ++j) {
      float z = sv + tvec[j];
      z = (z > 0.f) ? z : 0.2f * z;
      z += ewaR[j];
      z = (sadR[j] > 0.f) ? z : -1e9f;
      ebuf[i][j] = z;
      mx = fmaxf(mx, z);
    }
    float sum = 0.f;
    for (int j = 0; j < NN; ++j) {
      float p = expf(ebuf[i][j] - mx);
      ebuf[i][j] = p;
      sum += p;
    }
    float inv = 1.f / sum;
    for (int j0 = 0; j0 < NN; j0 += 8) {
      v8bf pk;
#pragma unroll
      for (int u = 0; u < 8; ++u) pk[u] = (bf16)(ebuf[i][j0 + u] * inv);
      *(v8bf*)&attb[i][j0] = pk;
    }
  }
  __syncthreads();
  int wave = tid >> 5, lane = tid & 31;
  int half = lane >> 4, nl = lane & 15;
  int mt = wave;                         // 8 waves -> 8 tile rows, 1 col tile (DH)
  const bf16* hpB = hpPack + (size_t)b * NN * HD;
  v8f acc = {};
#pragma unroll
  for (int k = 0; k < NN; k += 32) {
    const bf16* ap = &attb[mt * 16 + nl][k + half * 8];
    v16bf a = combine16(*(const v8bf*)ap, *(const v8bf*)(ap + 16));
    v16bf bfr = load_fragB_packed(hpB, HD, h * DHD, k, lane);
    acc = __builtin_amdgcn_wmma_f32_16x16x32_bf16(false, a, false, bfr, (short)0, acc,
                                                  false, false);
  }
#pragma unroll
  for (int v = 0; v < 8; ++v) {
    int m = mt * 16 + v + 8 * half;
    strong[((size_t)b * NN + m) * HD + h * DHD + nl] = acc[v];
  }
}

// per batch: weak = Aw @ hp[b] (WMMA, A-frag reused across all 8 n-tiles),
// g = relu(a*strong+(1-a)*weak), LayerNorm -> g_bf16 (row-major, A of FFN1)
__global__ __launch_bounds__(256)
void weak_combine_kernel(const bf16* __restrict__ awb, const bf16* __restrict__ hpPack,
                         const float* __restrict__ strong, const float* __restrict__ lng,
                         const float* __restrict__ lnb, bf16* __restrict__ gb) {
  __shared__ __align__(16) float gbuf[NN][HD];  // 64 KB
  int b = blockIdx.x;
  int tid = threadIdx.x, wave = tid >> 5, lane = tid & 31;
  int half = lane >> 4, nl = lane & 15;
  const bf16* hpB = hpPack + (size_t)b * NN * HD;
  int mt = wave;
  v8f acc[8];
#pragma unroll
  for (int q = 0; q < 8; ++q) { v8f z = {}; acc[q] = z; }
#pragma unroll
  for (int k = 0; k < HD; k += 32) {
    v16bf a = load_fragA(awb, NN, mt * 16, k, lane);
#pragma unroll
    for (int q = 0; q < 8; ++q) {
      v16bf bf = load_fragB_packed(hpB, HD, q * 16, k, lane);
      acc[q] = __builtin_amdgcn_wmma_f32_16x16x32_bf16(false, a, false, bf, (short)0,
                                                       acc[q], false, false);
    }
  }
#pragma unroll
  for (int q = 0; q < 8; ++q) {
    int n = q * 16 + nl;
#pragma unroll
    for (int v = 0; v < 8; ++v) {
      int m = mt * 16 + v + 8 * half;
      float sv  = strong[((size_t)b * NN + m) * HD + n];
      gbuf[m][n] = fmaxf(ALPHA_ * sv + (1.f - ALPHA_) * acc[q][v], 0.f);
    }
  }
  __syncthreads();
  if (tid < NN) {
    int m = tid;
    float mean = 0.f;
    for (int d = 0; d < HD; ++d) mean += gbuf[m][d];
    mean *= (1.f / HD);
    float var = 0.f;
    for (int d = 0; d < HD; ++d) { float df = gbuf[m][d] - mean; var += df * df; }
    var *= (1.f / HD);
    float inv = rsqrtf(var + EPS_);
    bf16* grow = gb + ((size_t)b * NN + m) * HD;
    for (int d0 = 0; d0 < HD; d0 += 8) {
      v8bf pk;
#pragma unroll
      for (int u = 0; u < 8; ++u)
        pk[u] = (bf16)((gbuf[m][d0 + u] - mean) * inv * lng[d0 + u] + lnb[d0 + u]);
      *(v8bf*)(grow + d0) = pk;
    }
  }
}

// final LayerNorm over HD per row -> bf16 (A operand of projection GEMM)
__global__ void ln_final_kernel(const float* __restrict__ h, const float* __restrict__ gma,
                                const float* __restrict__ bta, bf16* __restrict__ outb) {
  __shared__ float red[HD];
  int r = blockIdx.x, d = threadIdx.x;
  float xv = h[(size_t)r * HD + d];
  red[d] = xv;
  __syncthreads();
  for (int s = 64; s > 0; s >>= 1) { if (d < s) red[d] += red[d + s]; __syncthreads(); }
  float mean = red[0] * (1.f / HD);
  __syncthreads();
  float df = xv - mean;
  red[d] = df * df;
  __syncthreads();
  for (int s = 64; s > 0; s >>= 1) { if (d < s) red[d] += red[d + s]; __syncthreads(); }
  float var = red[0] * (1.f / HD);
  outb[(size_t)r * HD + d] = (bf16)(df * rsqrtf(var + EPS_) * gma[d] + bta[d]);
}

// mean-pool over N + 2-layer head -> out[B,60]
__global__ void head_kernel(const float* __restrict__ proj, const float* __restrict__ w1,
                            const float* __restrict__ b1, const float* __restrict__ w2,
                            const float* __restrict__ b2, float* __restrict__ out) {
  __shared__ float pooled[HD];
  __shared__ float hid[HD / 2];
  int b = blockIdx.x, t = threadIdx.x;
  float acc = 0.f;
  for (int n = 0; n < NN; ++n) acc += proj[((size_t)b * NN + n) * HD + t];
  pooled[t] = acc * (1.f / NN);
  __syncthreads();
  if (t < HD / 2) {
    float a = b1[t];
    for (int d = 0; d < HD; ++d) a += pooled[d] * w1[d * (HD / 2) + t];
    hid[t] = fmaxf(a, 0.f);
  }
  __syncthreads();
  if (t < OUTD) {
    float a = b2[t];
    for (int k = 0; k < HD / 2; ++k) a += hid[k] * w2[k * OUTD + t];
    out[(size_t)b * OUTD + t] = a;
  }
}

// ---------------- host launch ----------------
extern "C" void kernel_launch(void* const* d_in, const int* in_sizes, int n_in,
                              void* d_out, int out_size, void* d_ws, size_t ws_size,
                              hipStream_t stream) {
  const float* x      = (const float*)d_in[0];
  const float* conv_w = (const float*)d_in[1];
  const float* conv_b = (const float*)d_in[2];
  const float* bn_g   = (const float*)d_in[3];
  const float* bn_b   = (const float*)d_in[4];
  const float* bn_m   = (const float*)d_in[5];
  const float* bn_v   = (const float*)d_in[6];
  const float* W      = (const float*)d_in[7];
  const float* a_src  = (const float*)d_in[8];
  const float* a_dst  = (const float*)d_in[9];
  const float* ln_g   = (const float*)d_in[10];
  const float* ln_b   = (const float*)d_in[11];
  const float* ffn_w1 = (const float*)d_in[12];
  const float* ffn_b1 = (const float*)d_in[13];
  const float* ffn_w2 = (const float*)d_in[14];
  const float* ffn_b2 = (const float*)d_in[15];
  const float* norm_g = (const float*)d_in[16];
  const float* norm_b = (const float*)d_in[17];
  const float* proj_w = (const float*)d_in[18];
  const float* proj_b = (const float*)d_in[19];
  const float* head_w1 = (const float*)d_in[20];
  const float* head_b1 = (const float*)d_in[21];
  const float* head_w2 = (const float*)d_in[22];
  const float* head_b2 = (const float*)d_in[23];
  const float* strong_adj = (const float*)d_in[24];
  const float* weak_mask  = (const float*)d_in[25];
  const float* weak_param = (const float*)d_in[26];
  const float* edge_w_adj = (const float*)d_in[27];
  float* out = (float*)d_out;

  char* wptr = (char*)d_ws;
  auto alloc = [&](size_t bytes) -> char* {
    char* p = wptr;
    wptr += (bytes + 255) & ~(size_t)255;
    return p;
  };
  float* h_f32   = (float*)alloc((size_t)MM * HD * 4);
  bf16*  h_bf16  = (bf16*)alloc((size_t)MM * HD * 2);
  float* hp_f32  = (float*)alloc((size_t)MM * HD * 4);
  bf16*  hpPack  = (bf16*)alloc((size_t)MM * HD * 2);   // per-batch packed-B layout
  float* s_buf   = (float*)alloc((size_t)MM * NHD * 4);
  float* t_buf   = (float*)alloc((size_t)MM * NHD * 4);
  float* strong  = (float*)alloc((size_t)MM * HD * 4);
  bf16*  g_bf16  = (bf16*)alloc((size_t)MM * HD * 2);
  bf16*  mid_b   = (bf16*)alloc((size_t)MM * 2 * HD * 2);
  bf16*  hn_b    = (bf16*)alloc((size_t)MM * HD * 2);
  float* proj_o  = (float*)alloc((size_t)MM * HD * 4);
  bf16*  aw_b    = (bf16*)alloc((size_t)NN * NN * 2);
  bf16*  Wp      = (bf16*)alloc((size_t)LL * HD * HD * 2);
  bf16*  W1p     = (bf16*)alloc((size_t)LL * HD * 2 * HD * 2);
  bf16*  W2p     = (bf16*)alloc((size_t)LL * 2 * HD * HD * 2);
  bf16*  PWp     = (bf16*)alloc((size_t)HD * HD * 2);

  for (int i = 0; i < LL; ++i) {
    pack_b_kernel<<<64, 256, 0, stream>>>(W + (size_t)i * HD * HD,
                                          Wp + (size_t)i * HD * HD, HD, HD);
    pack_b_kernel<<<128, 256, 0, stream>>>(ffn_w1 + (size_t)i * HD * 2 * HD,
                                           W1p + (size_t)i * HD * 2 * HD, HD, 2 * HD);
    pack_b_kernel<<<128, 256, 0, stream>>>(ffn_w2 + (size_t)i * 2 * HD * HD,
                                           W2p + (size_t)i * 2 * HD * HD, 2 * HD, HD);
  }
  pack_b_kernel<<<64, 256, 0, stream>>>(proj_w, PWp, HD, HD);
  prep_aw_kernel<<<NN, NN, 0, stream>>>(weak_param, weak_mask, aw_b);

  embed_kernel<<<(MM * HD + 255) / 256, 256, 0, stream>>>(x, conv_w, conv_b, bn_g, bn_b,
                                                          bn_m, bn_v, h_f32, h_bf16);

  for (int i = 0; i < LL; ++i) {
    // hp = h @ W[i]  -> fp32 (for s/t) + per-batch packed-B (for attn/weak)
    gemm_kernel<HD, HD, 4, false, false, false, true, false, true>
        <<<dim3(MM / 128, HD / 64), 256, 0, stream>>>(
            h_bf16, Wp + (size_t)i * HD * HD, nullptr, nullptr, hp_f32, nullptr,
            hpPack);
    st_kernel<<<(MM * NHD + 255) / 256, 256, 0, stream>>>(
        hp_f32, a_src + (size_t)i * NHD * DHD, a_dst + (size_t)i * NHD * DHD,
        s_buf, t_buf);
    attn_strong_kernel<<<dim3(BB, NHD), 256, 0, stream>>>(
        s_buf, t_buf, edge_w_adj, strong_adj, hpPack, strong);
    weak_combine_kernel<<<BB, 256, 0, stream>>>(
        aw_b, hpPack, strong, ln_g + (size_t)i * HD, ln_b + (size_t)i * HD, g_bf16);
    // FFN1: relu(g @ w1 + b1) -> mid (row-major bf16 = A of FFN2)
    gemm_kernel<HD, 2 * HD, 4, true, false, true, false, true, false>
        <<<dim3(MM / 128, (2 * HD) / 64), 256, 0, stream>>>(
            g_bf16, W1p + (size_t)i * HD * 2 * HD, ffn_b1 + (size_t)i * 2 * HD,
            nullptr, nullptr, mid_b, nullptr);
    // FFN2: mid @ w2 + b2 + residual(h) -> h (fp32 + row-major bf16)
    gemm_kernel<2 * HD, HD, 4, true, true, false, true, true, false>
        <<<dim3(MM / 128, HD / 64), 256, 0, stream>>>(
            mid_b, W2p + (size_t)i * 2 * HD * HD, ffn_b2 + (size_t)i * HD,
            h_f32, h_f32, h_bf16, nullptr);
  }

  ln_final_kernel<<<MM, HD, 0, stream>>>(h_f32, norm_g, norm_b, hn_b);
  gemm_kernel<HD, HD, 4, true, false, false, true, false, false>
      <<<dim3(MM / 128, HD / 64), 256, 0, stream>>>(
          hn_b, PWp, proj_b, nullptr, proj_o, nullptr, nullptr);
  head_kernel<<<BB, HD, 0, stream>>>(proj_o, head_w1, head_b1, head_w2, head_b2, out);

  (void)in_sizes; (void)n_in; (void)out_size; (void)ws_size;
}
